// DotAttentionX_72378788872874
// MI455X (gfx1250) — matrix operand order repaired and checked by reading, not compiled, not run on hardware
//
#include <hip/hip_runtime.h>
#include <hip/hip_bf16.h>
#include <math.h>

typedef __attribute__((ext_vector_type(16))) _Float16 v16h;
typedef __attribute__((ext_vector_type(8)))  float    v8f;
typedef int v4i __attribute__((vector_size(16)));

#define L_SZ 2048
#define S_SZ 2048
#define D_SZ 64
#define BH_SZ 16   // B*H = 2*8

// ---------------------------------------------------------------------------
// CDNA5 async global->LDS helpers (ASYNCcnt path), builtin with asm fallback
// ---------------------------------------------------------------------------
typedef __attribute__((address_space(1))) v4i* as1_v4ip;
typedef __attribute__((address_space(3))) v4i* as3_v4ip;
typedef __attribute__((address_space(3))) void* as3_vp;

__device__ __forceinline__ void async_b128_to_lds(const void* gsrc, void* ldst)
{
#if __has_builtin(__builtin_amdgcn_global_load_async_to_lds_b128)
    __builtin_amdgcn_global_load_async_to_lds_b128((as1_v4ip)gsrc, (as3_v4ip)ldst,
                                                   0, 0);
#else
    unsigned int loff = (unsigned int)(unsigned long long)(as3_vp)ldst;
    asm volatile("global_load_async_to_lds_b128 %0, %1, off"
                 :: "v"(loff), "v"(gsrc) : "memory");
#endif
}

__device__ __forceinline__ void wait_async0()
{
#if __has_builtin(__builtin_amdgcn_s_wait_asynccnt)
    __builtin_amdgcn_s_wait_asynccnt(0);
#else
    asm volatile("s_wait_asynccnt 0x0" ::: "memory");
#endif
}

__device__ __forceinline__ float fast_exp2(float x)
{
#if __has_builtin(__builtin_amdgcn_exp2f)
    return __builtin_amdgcn_exp2f(x);      // v_exp_f32 (natively 2^x)
#else
    return exp2f(x);
#endif
}

// ---------------------------------------------------------------------------
// Kernel 1: row sums of query and key over D=64 (qsum[bh,l], ksum[bh,s])
// ---------------------------------------------------------------------------
__global__ __launch_bounds__(256) void rowsum_kernel(
    const float* __restrict__ q, const float* __restrict__ k,
    float* __restrict__ qs, float* __restrict__ ks)
{
    int r = blockIdx.x * 256 + threadIdx.x;         // 0 .. 65535
    const float* src;
    float* dst;
    int rr;
    if (r < BH_SZ * L_SZ) { src = q; dst = qs; rr = r; }
    else                  { src = k; dst = ks; rr = r - BH_SZ * L_SZ; }
    const float4* p = (const float4*)(src + (size_t)rr * D_SZ);
    float s = 0.f;
#pragma unroll
    for (int i = 0; i < 16; ++i) {
        float4 v = p[i];
        s += v.x + v.y + v.z + v.w;
    }
    dst[rr] = s;
}

// ---------------------------------------------------------------------------
// Kernel 2: per (b,h) max/min of ksum  (gives exact softmax row max, rank-1)
// ---------------------------------------------------------------------------
__global__ __launch_bounds__(256) void kminmax_kernel(
    const float* __restrict__ ks, float* __restrict__ kmx, float* __restrict__ kmn)
{
    __shared__ float smx[256], smn[256];
    int bh = blockIdx.x, tid = threadIdx.x;
    float mx = -INFINITY, mn = INFINITY;
    for (int i = tid; i < S_SZ; i += 256) {
        float v = ks[bh * S_SZ + i];
        mx = fmaxf(mx, v);
        mn = fminf(mn, v);
    }
    smx[tid] = mx; smn[tid] = mn;
    __syncthreads();
    for (int st = 128; st > 0; st >>= 1) {
        if (tid < st) {
            smx[tid] = fmaxf(smx[tid], smx[tid + st]);
            smn[tid] = fminf(smn[tid], smn[tid + st]);
        }
        __syncthreads();
    }
    if (tid == 0) { kmx[bh] = smx[0]; kmn[bh] = smn[0]; }
}

// ---------------------------------------------------------------------------
// Kernel 3: convert value f32 [BH,S,D] -> f16 transposed [BH,D,S]
// so WMMA B-fragments (fixed d, 16 consecutive s) are contiguous 32B loads.
// ---------------------------------------------------------------------------
__global__ __launch_bounds__(256) void vtrans_kernel(
    const float* __restrict__ v, _Float16* __restrict__ vt)
{
    int i = blockIdx.x * 256 + threadIdx.x;          // 0 .. BH*S*D-1
    int d  = i & (D_SZ - 1);
    int s  = (i >> 6) & (S_SZ - 1);
    int bh = i >> 17;
    vt[((size_t)bh * D_SZ + d) * S_SZ + s] = (_Float16)v[i];
}

// ---------------------------------------------------------------------------
// Kernel 4: main P@V with on-the-fly P = exp2(a2*ksum - m2), WMMA f16->f32.
// Block = 256 threads = 8 waves; each wave owns 16 L-rows, all 64 d-cols.
// V tiles (32 s x 64 d) are double-buffered in LDS via async global->LDS
// copies (ASYNCcnt): one barrier per chunk, next tile's copy overlaps the
// exp/WMMA body of the current chunk.
// ---------------------------------------------------------------------------
__global__ __launch_bounds__(256, 1) void attn_main_kernel(
    const float*    __restrict__ qsum,   // [BH, L]
    const float*    __restrict__ ksum,   // [BH, S]
    const float*    __restrict__ kmx,    // [BH]
    const float*    __restrict__ kmn,    // [BH]
    const _Float16* __restrict__ vt,     // [BH, D, S]
    float*          __restrict__ out)    // [BH, L, D]
{
    __shared__ float    lds_ks[S_SZ];             // 8 KB: ksum for this (b,h)
    __shared__ _Float16 lds_v[2][32 * D_SZ];      // 2 x 4 KB: V tiles [d][32 s]

    const int tid  = threadIdx.x;
    const int lane = tid & 31;
    const int wave = tid >> 5;                 // 0..7
    const int bh   = blockIdx.y;               // 0..15
    const int l0   = blockIdx.x * 128 + wave * 16;

    // Preload ksum for this (b,h) into LDS (coalesced).
    for (int i = tid; i < S_SZ; i += 256)
        lds_ks[i] = ksum[bh * S_SZ + i];

    // Per-row scalar a and exact softmax max m (rank-1 score), in log2 units.
    const int   row = l0 + (lane & 15);        // lanes n and n+16 share a row
    const float a   = qsum[bh * L_SZ + row];
    const float m   = (a >= 0.f) ? a * kmx[bh] : a * kmn[bh];
    const float LOG2E = 1.4426950408889634f;
    const float a2  = a * LOG2E;
    const float m2  = m * LOG2E;

    // A fragment, contiguous k order chosen here (B matches): lane<16 holds K {0..15}, lane>=16 holds
    // K {16..31} of its row.
    const int kbase = (lane & 16);             // 0 or 16
    const int sboff = (lane & 16);             // B-fragment s offset: 0 or 16

    const size_t vbase = (size_t)bh * D_SZ * S_SZ;
    const int vld_d = tid >> 2;                // cooperative tile loader coords
    const int vld_q = tid & 3;
    const _Float16* gv = vt + vbase + (size_t)vld_d * S_SZ + vld_q * 8;

    v8f acc0 = {}, acc1 = {}, acc2 = {}, acc3 = {};
    float zpart = 0.f;

    // Kick off tile 0 while ksum preload is in flight.
    async_b128_to_lds(gv, &lds_v[0][vld_d * 32 + vld_q * 8]);

    for (int it = 0; it < S_SZ / 32; ++it) {
        const int s0  = it * 32;
        const int buf = it & 1;

        wait_async0();                 // this thread's tile `it` landed
        __syncthreads();               // everyone's tile `it` + lds_ks visible

        if (it + 1 < S_SZ / 32)        // async-copy next tile into other buffer
            async_b128_to_lds(gv + s0 + 32,
                              &lds_v[buf ^ 1][vld_d * 32 + vld_q * 8]);

        // Build P fragment: 16 values per lane, f32 partial Z, f16 for WMMA.
        float kv[16];
        *(float4*)&kv[0]  = *(const float4*)(&lds_ks[s0 + kbase]);
        *(float4*)&kv[4]  = *(const float4*)(&lds_ks[s0 + kbase + 4]);
        *(float4*)&kv[8]  = *(const float4*)(&lds_ks[s0 + kbase + 8]);
        *(float4*)&kv[12] = *(const float4*)(&lds_ks[s0 + kbase + 12]);

        v16h pa;
#pragma unroll
        for (int i = 0; i < 16; ++i) {
            float p = fast_exp2(fmaf(a2, kv[i], -m2));  // v_exp_f32 (TRANS)
            zpart += p;
            pa[i] = (_Float16)p;
        }

        // B fragments from LDS: lane n = column d, 16 contiguous K halfs.
        const int n = lane & 15;
        const _Float16* tb = &lds_v[buf][0];
        v16h b0 = *(const v16h*)(&tb[( 0 + n) * 32 + sboff]);
        v16h b1 = *(const v16h*)(&tb[(16 + n) * 32 + sboff]);
        v16h b2 = *(const v16h*)(&tb[(32 + n) * 32 + sboff]);
        v16h b3 = *(const v16h*)(&tb[(48 + n) * 32 + sboff]);

        acc0 = __builtin_amdgcn_wmma_f32_16x16x32_f16(false, pa, false, b0,
                                                      (short)0, acc0, false, false);
        acc1 = __builtin_amdgcn_wmma_f32_16x16x32_f16(false, pa, false, b1,
                                                      (short)0, acc1, false, false);
        acc2 = __builtin_amdgcn_wmma_f32_16x16x32_f16(false, pa, false, b2,
                                                      (short)0, acc2, false, false);
        acc3 = __builtin_amdgcn_wmma_f32_16x16x32_f16(false, pa, false, b3,
                                                      (short)0, acc3, false, false);
    }

    // Z per row: halves live in lanes n and n+16.
    float z = zpart + __shfl_xor(zpart, 16, 32);

    // C/D layout: lane<16 element r -> row l0+r; lane>=16 -> row l0+8+r.
    const int n     = lane & 15;
    const int rbase = (lane & 16) ? 8 : 0;
#pragma unroll
    for (int r = 0; r < 8; ++r) {
        float zr  = __shfl(z, rbase + r, 32);   // lane rbase+r holds that row's Z
        float inv = 1.0f / zr;
        float* op = out + ((size_t)bh * L_SZ + l0 + rbase + r) * D_SZ + n;
        op[ 0] = acc0[r] * inv;
        op[16] = acc1[r] * inv;
        op[32] = acc2[r] * inv;
        op[48] = acc3[r] * inv;
    }
}

// ---------------------------------------------------------------------------
extern "C" void kernel_launch(void* const* d_in, const int* in_sizes, int n_in,
                              void* d_out, int out_size, void* d_ws, size_t ws_size,
                              hipStream_t stream)
{
    const float* q = (const float*)d_in[0];
    const float* k = (const float*)d_in[1];
    const float* v = (const float*)d_in[2];
    float* out = (float*)d_out;

    // Workspace layout (~4.3 MB total)
    float* qs  = (float*)d_ws;                      // 32768 f32
    float* ks  = qs + BH_SZ * L_SZ;                 // 32768 f32
    float* kmx = ks + BH_SZ * S_SZ;                 // 16 f32
    float* kmn = kmx + BH_SZ;                       // 16 f32
    _Float16* vt = (_Float16*)((char*)d_ws + 272 * 1024);   // BH*D*S halfs = 4 MB

    (void)in_sizes; (void)n_in; (void)out_size; (void)ws_size;

    // 1) row sums of q and k
    rowsum_kernel<<<(BH_SZ * (L_SZ + S_SZ)) / 256, 256, 0, stream>>>(q, k, qs, ks);
    // 2) per-(b,h) max/min of ksum
    kminmax_kernel<<<BH_SZ, 256, 0, stream>>>(ks, kmx, kmn);
    // 3) value -> f16, transposed [BH, D, S]
    vtrans_kernel<<<(BH_SZ * S_SZ * D_SZ) / 256, 256, 0, stream>>>(v, vt);
    // 4) main WMMA attention
    dim3 grid(L_SZ / 128, BH_SZ);
    attn_main_kernel<<<grid, 256, 0, stream>>>(qs, ks, kmx, kmn, vt, out);
}
